// Offset2D_43190191129117
// MI455X (gfx1250) — matrix operand order repaired and checked by reading, not compiled, not run on hardware
//
#include <hip/hip_runtime.h>

// Problem constants (from reference setup_inputs)
#define B_   4
#define C_   64
#define H_   512
#define W_   512
#define HW_  (H_ * W_)
#define EPS_ 1e-5f

typedef __attribute__((ext_vector_type(16))) _Float16 v16h;
typedef __attribute__((ext_vector_type(8)))  float    v8f;
typedef __attribute__((ext_vector_type(4)))  float    v4f;

// ---------------------------------------------------------------------------
// Kernel 0: zero the fa accumulator (lives in d_out) and set aa = EPS (ws).
// ---------------------------------------------------------------------------
__global__ void offset2d_init(float* __restrict__ fa, float* __restrict__ aa) {
    size_t tid    = (size_t)blockIdx.x * blockDim.x + threadIdx.x;
    size_t stride = (size_t)gridDim.x * blockDim.x;
    size_t n4 = ((size_t)B_ * C_ * HW_) / 4;
    v4f z = {0.f, 0.f, 0.f, 0.f};
    for (size_t i = tid; i < n4; i += stride) ((v4f*)fa)[i] = z;
    for (size_t i = tid; i < (size_t)B_ * HW_; i += stride) aa[i] = EPS_;
}

// ---------------------------------------------------------------------------
// Kernel 1 (fused): 1x1 conv via WMMA + offsets/att/index + atomic scatter.
// Each wave handles a tile of 16 pixels:
//   A (16x32 f16): w[3,64] split into two K-chunks, rows 3..15 zero.
//   B (32x16 f16): 32 channels x 16 pixels of x (converted from the fp32
//                  loads that are also kept for the full-precision scatter).
//   D (16x16 f32): rows 0..2 = oa0/oa1/oa2 for the 16 pixel columns.
// ---------------------------------------------------------------------------
__global__ void offset2d_main(const float* __restrict__ x,
                              const float* __restrict__ w,
                              const float* __restrict__ bias,
                              float* __restrict__ fa,       // d_out: [B,C,HW]
                              float* __restrict__ off_out,  // d_out: [B,2,HW]
                              float* __restrict__ dst_out,  // d_out: [B,2,HW]
                              float* __restrict__ aa)       // ws:    [B,HW]
{
    const int lane   = threadIdx.x & 31;                 // gfx1250 is wave32
    const int wid    = ((int)blockIdx.x * (int)blockDim.x + (int)threadIdx.x) >> 5;
    const int nwaves = ((int)gridDim.x * (int)blockDim.x) >> 5;
    const bool hi    = lane >= 16;
    const int  n     = lane & 15;                        // pixel column / M row
    const int  cOff  = hi ? 16 : 0;

    // --- Build A operands once (weights are tiny, L2-resident) -------------
    // A 16x32 order used here (contiguous k, matched by B): lanes 0-15 hold M=lane, K = {0..15};
    // lanes 16-31 hold M=lane-16, K = {16..31} (2 halves per VGPR).
    v16h a0, a1;
    #pragma unroll
    for (int v = 0; v < 8; ++v) {
        #pragma unroll
        for (int h2 = 0; h2 < 2; ++h2) {
            int K = 2 * v + h2 + (hi ? 16 : 0);
            float e0 = (n < 3) ? w[n * 64 + K]      : 0.0f;
            float e1 = (n < 3) ? w[n * 64 + 32 + K] : 0.0f;
            a0[2 * v + h2] = (_Float16)e0;
            a1[2 * v + h2] = (_Float16)e1;
        }
    }
    const float b0 = bias[0], b1 = bias[1], b2 = bias[2];

    const int totalTiles = (B_ * HW_) / 16;
    for (int t = wid; t < totalTiles; t += nwaves) {
        const int pbase = t * 16;                 // flat index b*HW + p0
        const int b     = pbase / HW_;
        const int p     = (pbase - b * HW_) + n;  // this lane's pixel in image
        const float* xb = x + (size_t)b * C_ * HW_ + (size_t)(pbase - b * HW_) + n;

        // Prefetch next tile's first lines into cache (global_prefetch_b8).
        if (t + nwaves < totalTiles) {
            const int pb2 = (t + nwaves) * 16;
            const int b2  = pb2 / HW_;
            __builtin_prefetch(x + (size_t)b2 * C_ * HW_ + (pb2 - b2 * HW_) + n, 0, 1);
        }

        // --- Load 32 fp32 channels per lane (kept for exact scatter) -------
        // B 32x16 order used here (same as A): lanes 0-15 K=0..15, lanes 16-31 K=16..31, so this
        // lane owns channels {k0 + cOff + j : j=0..15} for k0 in {0,32}.
        float xv0[16], xv1[16];
        #pragma unroll
        for (int j = 0; j < 16; ++j) xv0[j] = xb[(size_t)(cOff + j) * HW_];
        #pragma unroll
        for (int j = 0; j < 16; ++j) xv1[j] = xb[(size_t)(32 + cOff + j) * HW_];

        v16h bm0, bm1;
        #pragma unroll
        for (int j = 0; j < 16; ++j) {
            bm0[j] = (_Float16)xv0[j];
            bm1[j] = (_Float16)xv1[j];
        }

        v8f d = {};
        d = __builtin_amdgcn_wmma_f32_16x16x32_f16(false, a0, false, bm0,
                                                   (short)0, d, false, false);
        d = __builtin_amdgcn_wmma_f32_16x16x32_f16(false, a1, false, bm1,
                                                   (short)0, d, false, false);

        // Lanes 0..15: d[0..2] = oa rows 0..2 for pixel column n. Hi lanes
        // see zero rows (A rows >= 3 are zero) -> computed but unused.
        float offy = (d[0] + b0) * (float)H_;     // DOWNSAMPLE == 1.0
        float offx = (d[1] + b1) * (float)W_;
        float att  = __expf(d[2] + b2);
        int gy = p >> 9;                          // p / W (W = 512)
        int gx = p & (W_ - 1);
        int dy = (int)rintf((float)gy + offy);    // round-half-even == jnp.round
        int dx = (int)rintf((float)gx + offx);
        dy = dy < 0 ? 0 : (dy > H_ - 1 ? H_ - 1 : dy);
        dx = dx < 0 ? 0 : (dx > W_ - 1 ? W_ - 1 : dx);
        int idx = dy * W_ + dx;

        if (!hi) {
            off_out[((size_t)b * 2    ) * HW_ + p] = offy;
            off_out[((size_t)b * 2 + 1) * HW_ + p] = offx;
            dst_out[((size_t)b * 2    ) * HW_ + p] = (float)dy;
            dst_out[((size_t)b * 2 + 1) * HW_ + p] = (float)dx;
            atomicAdd(&aa[(size_t)b * HW_ + idx], att);
        }

        // Broadcast att/idx from the owning low lane to its hi partner so
        // all 32 lanes scatter their 32 channels (64 ch total per pixel).
        float attb = __shfl(att, n, 32);
        int   idxb = __shfl(idx, n, 32);

        float* fab = fa + (size_t)b * C_ * HW_ + idxb;
        #pragma unroll
        for (int j = 0; j < 16; ++j)
            atomicAdd(&fab[(size_t)(cOff + j) * HW_], xv0[j] * attb);
        #pragma unroll
        for (int j = 0; j < 16; ++j)
            atomicAdd(&fab[(size_t)(32 + cOff + j) * HW_], xv1[j] * attb);
    }
}

// ---------------------------------------------------------------------------
// Kernel 2: out = fa / aa, in place in d_out. One thread per (b, pixel),
// sweeping 64 channels (coalesced across lanes; aa loaded once per pixel).
// ---------------------------------------------------------------------------
__global__ void offset2d_norm(float* __restrict__ fa, const float* __restrict__ aa) {
    int tid    = (int)blockIdx.x * (int)blockDim.x + (int)threadIdx.x;
    int stride = (int)gridDim.x * (int)blockDim.x;
    for (int i = tid; i < B_ * HW_; i += stride) {
        int b = i / HW_;
        int p = i - b * HW_;
        float r = 1.0f / aa[i];
        float* f = fa + (size_t)b * C_ * HW_ + p;
        #pragma unroll
        for (int c = 0; c < C_; ++c) f[(size_t)c * HW_] *= r;
    }
}

// ---------------------------------------------------------------------------
extern "C" void kernel_launch(void* const* d_in, const int* in_sizes, int n_in,
                              void* d_out, int out_size, void* d_ws, size_t ws_size,
                              hipStream_t stream) {
    const float* x    = (const float*)d_in[0];   // [B,C,H,W] fp32
    const float* w    = (const float*)d_in[1];   // [3,C]     fp32
    const float* bias = (const float*)d_in[2];   // [3]       fp32

    float* out     = (float*)d_out;
    float* fa      = out;                                    // [B,C,HW]
    float* off_out = out + (size_t)B_ * C_ * HW_;            // [B,2,HW]
    float* dst_out = off_out + (size_t)B_ * 2 * HW_;         // [B,2,HW]
    float* aa      = (float*)d_ws;                           // [B,HW] (4 MB)

    offset2d_init<<<2048, 256, 0, stream>>>(fa, aa);
    // 8192 blocks * 8 waves = 65536 waves -> one 16-pixel tile per wave.
    offset2d_main<<<8192, 256, 0, stream>>>(x, w, bias, fa, off_out, dst_out, aa);
    offset2d_norm<<<2048, 256, 0, stream>>>(fa, aa);
}
